// PointConvTranspose_21294447854193
// MI455X (gfx1250) — matrix operand order repaired and checked by reading, not compiled, not run on hardware
//
#include <hip/hip_runtime.h>

// ---------------- CDNA5 WMMA / TDM types ----------------
typedef __attribute__((ext_vector_type(16))) __bf16 v16bf;
typedef __attribute__((ext_vector_type(8)))  float  v8f;
typedef __attribute__((ext_vector_type(4)))  unsigned v4u;
typedef __attribute__((ext_vector_type(8)))  int      v8i;
typedef __attribute__((ext_vector_type(4)))  int      v4i;

union BFVec { unsigned u[8]; v16bf v; };

__device__ __forceinline__ unsigned short f2bf(float x) {
  unsigned u = __float_as_uint(x);
  unsigned r = 0x7FFFu + ((u >> 16) & 1u);
  return (unsigned short)((u + r) >> 16);
}
__device__ __forceinline__ unsigned pk2bf(float a, float b) {
  return (unsigned)f2bf(a) | ((unsigned)f2bf(b) << 16);
}

// ---------------- problem constants ----------------
#define PTS   16    // dense points per block (= WMMA M of final GEMM)
#define KNN   16
#define CIN   128
#define PEC   32
#define MID   16
#define CFEAT 160   // CIN + PEC
#define COUT  128
#define KTOT  2560  // CFEAT * MID
#define KCH   64    // k per TDM chunk (2 WMMA k-steps)
#define NCH   (KTOT / KCH)   // 40 chunks

// ---- dynamic LDS byte offsets ----
#define OFF_LIN    0         // 384 f32: lin_b|lin_g|lin_be          1536 B
#define OFF_NEI    1536      // 256 int                              1024 B
#define OFF_PE     2560      // [16p][32c][16k] u16 (transposed)    16384 B
#define OFF_WTS    18944     // [16p][16m][16k] u16 (transposed)     8192 B
#define OFF_OUTER  27136     // [16p][2560] u16                     81920 B
#define OFF_OUTF   109056    // [16p][128] f32                       8192 B
#define OFF_RED    117248    // 16*16*2 f32                          2048 B
#define OFF_MU     119296    // 16 f32
#define OFF_RS     119360    // 16 f32
#define OFF_BB     119424    // TDM B double-buffers: 8 waves * 2 * 2048 B
#define SMEM_BYTES 152192
// MLP staging buffers live inside the (not-yet-used) sOuter region:
#define OFF_Y   (OFF_OUTER)            // [256 pair][32] f32  : 32768 B
#define OFF_H   (OFF_OUTER + 32768)    // [256 pair][32] u16  : 16384 B
#define OFF_W0  (OFF_OUTER + 49152)    // [256 pair][16] u16  :  8192 B
#define OFF_Z   (OFF_OUTER + 57344)    // [256 pair][16] f32  : 16384 B (ends 73728 < 81920)

template <int N>
__device__ __forceinline__ void lnact(float* x, const float* g, const float* b, bool act) {
  float mu = 0.f;
#pragma unroll
  for (int i = 0; i < N; i++) mu += x[i];
  mu *= (1.0f / N);
  float var = 0.f;
#pragma unroll
  for (int i = 0; i < N; i++) { float d = x[i] - mu; var = fmaf(d, d, var); }
  var *= (1.0f / N);
  float rs = rsqrtf(var + 1e-5f);
#pragma unroll
  for (int i = 0; i < N; i++) {
    float y = fmaf((x[i] - mu) * rs, g[i], b[i]);
    x[i] = act ? (y >= 0.f ? y : 0.1f * y) : y;
  }
}

// ---- Tensor Data Mover: 2-D tile load (bf16), per D# bit layout (ISA 8.3/8.4) ----
__device__ __forceinline__ void tdm_issue(unsigned ldsOff, const unsigned short* gptr) {
  unsigned long long ga = (unsigned long long)(uintptr_t)gptr;
  v4u g0;
  g0[0] = 1u;                                                  // count=1, user mode
  g0[1] = ldsOff;                                              // LDS byte address
  g0[2] = (unsigned)ga;                                        // global_addr[31:0]
  g0[3] = (unsigned)((ga >> 32) & 0x01FFFFFFu) | (2u << 30);   // addr[56:32] | type=2
  v8i g1;
  const unsigned dim0 = 1u << 20, dim1 = 1u << 16;             // tensor dims (no OOB)
  g1[0] = (int)(1u << 16);                                     // wg_mask=0, data_size=2B
  g1[1] = (int)((dim0 & 0xFFFFu) << 16);                       // tensor_dim0[15:0]
  g1[2] = (int)((dim0 >> 16) | ((dim1 & 0xFFFFu) << 16));      // dim0[31:16]|dim1[15:0]
  g1[3] = (int)((dim1 >> 16) | ((unsigned)KCH << 16));         // dim1[31:16]|tile_dim0
  g1[4] = (int)16u;                                            // tile_dim1=16
  g1[5] = (int)KTOT;                                           // tensor_dim0_stride
  g1[6] = 0; g1[7] = 0;
  v4i z4 = {0, 0, 0, 0};
#if defined(__clang_major__) && __clang_major__ >= 23
  v8i z8 = {0, 0, 0, 0, 0, 0, 0, 0};
  __builtin_amdgcn_tensor_load_to_lds(g0, g1, z4, z4, z8, 0);
#else
  __builtin_amdgcn_tensor_load_to_lds(g0, g1, z4, z4, 0);
#endif
}

// 16->16 layer (K=16, zero-padded to 32) as WMMA, two M-tiles per wave.
__device__ __forceinline__ void wn_layer_wmma(const unsigned short* wbufIn, float* zbufOut,
                                              const unsigned short* __restrict__ wT,
                                              int wv, int lane) {
  const int lrow = lane & 15;
  const bool lo = (lane < 16);
  const int rb = lo ? 0 : 8;
  BFVec Bw;  // B[k][col] = wT[col*16 + k]; lanes<16 carry K0..7, lanes>=16 K8..15; K16..31 = 0
  {
    const uint4* bp = (const uint4*)(wT + lrow * MID);
    uint4 b0 = bp[0], b1 = bp[1];
    Bw.u[0] = lo ? b0.x : b1.x; Bw.u[1] = lo ? b0.y : b1.y;
    Bw.u[2] = lo ? b0.z : b1.z; Bw.u[3] = lo ? b0.w : b1.w;
    Bw.u[4] = 0u; Bw.u[5] = 0u;
    Bw.u[6] = 0u; Bw.u[7] = 0u;
  }
#pragma unroll
  for (int mt = 0; mt < 2; ++mt) {
    int rowbase = wv * 32 + mt * 16;
    BFVec Aw;
    const uint4* aw = (const uint4*)(wbufIn + (rowbase + lrow) * MID + (lo ? 0 : 8));
    uint4 a0 = aw[0];
    Aw.u[0] = a0.x; Aw.u[1] = a0.y; Aw.u[2] = a0.z; Aw.u[3] = a0.w;
    Aw.u[4] = 0u; Aw.u[5] = 0u; Aw.u[6] = 0u; Aw.u[7] = 0u;
    v8f acc;
#pragma unroll
    for (int i = 0; i < 8; i++) acc[i] = 0.f;
    acc = __builtin_amdgcn_wmma_f32_16x16x32_bf16(false, Aw.v, false, Bw.v,
                                                  (short)0, acc, false, false);
#pragma unroll
    for (int r = 0; r < 8; r++)
      zbufOut[(rowbase + rb + r) * MID + lrow] = acc[r];
  }
}

// ---------------- pre-kernels: f32 -> bf16 conversions ----------------
__global__ __launch_bounds__(256) void cvt_linw(const float* __restrict__ w,
                                                unsigned short* __restrict__ wt) {
  int i = blockIdx.x * 256 + threadIdx.x;
  if (i < KTOT * COUT) {
    int k = i / COUT, co = i % COUT;
    wt[(size_t)co * KTOT + k] = f2bf(w[i]);   // col-major [co][k]
  }
}

__global__ __launch_bounds__(256) void cvt_bf(const float* __restrict__ f,
                                              unsigned short* __restrict__ o, int n) {
  int i = blockIdx.x * 256 + threadIdx.x;
  if (i < n) o[i] = f2bf(f[i]);
}

// transpose+convert the three small layer weights: [in][out] f32 -> [out][in] bf16
__global__ __launch_bounds__(256) void cvt_small(const float* __restrict__ pw1,
                                                 const float* __restrict__ ww1,
                                                 const float* __restrict__ ww2,
                                                 unsigned short* __restrict__ o) {
  int i = blockIdx.x * 256 + threadIdx.x;
  if (i < 1024)      { int oc = i / 32, ii = i % 32;            o[oc * 32 + ii]        = f2bf(pw1[ii * 32 + oc]); }
  else if (i < 1280) { int j = i - 1024; int oc = j / 16, ii = j % 16; o[1024 + oc * 16 + ii] = f2bf(ww1[ii * 16 + oc]); }
  else if (i < 1536) { int j = i - 1280; int oc = j / 16, ii = j % 16; o[1280 + oc * 16 + ii] = f2bf(ww2[ii * 16 + oc]); }
}

// ---------------- fused PointConvTranspose kernel ----------------
__global__ __launch_bounds__(256) void pointconv_main(
    const float* __restrict__ sxyz, const int* __restrict__ nei,
    const float* __restrict__ dxyz, const float* __restrict__ dfeats,
    const unsigned short* __restrict__ featsBF, const unsigned short* __restrict__ WtBF,
    const unsigned short* __restrict__ peW1T, const unsigned short* __restrict__ wnW1T,
    const unsigned short* __restrict__ wnW2T,
    const float* __restrict__ pe_w0, const float* __restrict__ pe_b0,
    const float* __restrict__ pe_g0, const float* __restrict__ pe_be0,
    const float* __restrict__ pe_b1, const float* __restrict__ pe_g1,
    const float* __restrict__ pe_be1,
    const float* __restrict__ wn_w0, const float* __restrict__ wn_b0,
    const float* __restrict__ wn_g0, const float* __restrict__ wn_be0,
    const float* __restrict__ wn_b1, const float* __restrict__ wn_g1,
    const float* __restrict__ wn_be1,
    const float* __restrict__ wn_b2, const float* __restrict__ wn_g2,
    const float* __restrict__ wn_be2,
    const float* __restrict__ lin_b, const float* __restrict__ lin_g,
    const float* __restrict__ lin_be,
    float* __restrict__ out, int Nd) {
  extern __shared__ char smem_raw[];
  float*          sLin   = (float*)(smem_raw + OFF_LIN);
  int*            sNei   = (int*)(smem_raw + OFF_NEI);
  unsigned short* sPeT   = (unsigned short*)(smem_raw + OFF_PE);
  unsigned short* sWtsT  = (unsigned short*)(smem_raw + OFF_WTS);
  unsigned short* sOuter = (unsigned short*)(smem_raw + OFF_OUTER);
  float*          sOutF  = (float*)(smem_raw + OFF_OUTF);
  float*          sRed   = (float*)(smem_raw + OFF_RED);
  float*          sMu    = (float*)(smem_raw + OFF_MU);
  float*          sRs    = (float*)(smem_raw + OFF_RS);
  float*          ybuf   = (float*)(smem_raw + OFF_Y);
  unsigned short* hbuf   = (unsigned short*)(smem_raw + OFF_H);
  unsigned short* wbuf   = (unsigned short*)(smem_raw + OFF_W0);
  float*          zbuf   = (float*)(smem_raw + OFF_Z);

  const int tid  = threadIdx.x;
  const int wv   = tid >> 5;          // wave 0..7 (wave32)
  const int lane = tid & 31;
  const int lrow = lane & 15;
  const bool lo  = (lane < 16);
  const int pair = tid;               // (point,neighbor) pair = p*16+kk
  const int p    = pair >> 4, kk = pair & 15;

  // ---- kick off TDM prefetch of the first two final-GEMM B chunks for this wave ----
  const int wvu = __builtin_amdgcn_readfirstlane(wv);
  const unsigned bbase = OFF_BB + (unsigned)wvu * 4096u;
  const unsigned short* wcol = WtBF + (size_t)(wvu * 16) * KTOT;
  tdm_issue(bbase,         wcol);          // chunk 0 -> buf 0
  tdm_issue(bbase + 2048u, wcol + KCH);    // chunk 1 -> buf 1

  for (int i = tid; i < 128; i += 256) {
    sLin[i]       = lin_b[i];
    sLin[128 + i] = lin_g[i];
    sLin[256 + i] = lin_be[i];
  }

  // ===== phase A: per-pair input layers (K=3, VALU) =====
  {
    int n = blockIdx.x * PTS + p;
    bool valid = (n < Nd);
    int nidx = valid ? nei[n * KNN + kk] : 0;
    sNei[tid] = nidx;
    float lx = 0.f, ly = 0.f, lz = 0.f;
    if (valid) {
      lx = sxyz[nidx * 3 + 0] - dxyz[n * 3 + 0];
      ly = sxyz[nidx * 3 + 1] - dxyz[n * 3 + 1];
      lz = sxyz[nidx * 3 + 2] - dxyz[n * 3 + 2];
    }
    float h[PEC];
#pragma unroll
    for (int o = 0; o < PEC; o++)
      h[o] = fmaf(lz, pe_w0[64 + o], fmaf(ly, pe_w0[32 + o], fmaf(lx, pe_w0[o], pe_b0[o])));
    lnact<PEC>(h, pe_g0, pe_be0, true);
    uint4* hv = (uint4*)(hbuf + pair * PEC);
#pragma unroll
    for (int j = 0; j < 4; ++j) {
      uint4 v;
      v.x = pk2bf(h[8 * j + 0], h[8 * j + 1]);
      v.y = pk2bf(h[8 * j + 2], h[8 * j + 3]);
      v.z = pk2bf(h[8 * j + 4], h[8 * j + 5]);
      v.w = pk2bf(h[8 * j + 6], h[8 * j + 7]);
      hv[j] = v;
    }
    float w0[MID];
#pragma unroll
    for (int o = 0; o < MID; o++)
      w0[o] = fmaf(lz, wn_w0[32 + o], fmaf(ly, wn_w0[16 + o], fmaf(lx, wn_w0[o], wn_b0[o])));
    lnact<MID>(w0, wn_g0, wn_be0, true);
    uint4* wv4 = (uint4*)(wbuf + pair * MID);
#pragma unroll
    for (int j = 0; j < 2; ++j) {
      uint4 v;
      v.x = pk2bf(w0[8 * j + 0], w0[8 * j + 1]);
      v.y = pk2bf(w0[8 * j + 2], w0[8 * j + 3]);
      v.z = pk2bf(w0[8 * j + 4], w0[8 * j + 5]);
      v.w = pk2bf(w0[8 * j + 6], w0[8 * j + 7]);
      wv4[j] = v;
    }
  }
  __syncthreads();

  // ===== phase B: PE-L1 [256x32 @ 32x32] + WN-L1 [256x16 @ 16x16] via WMMA =====
  {
    const int rb = lo ? 0 : 8;
#pragma unroll
    for (int mt = 0; mt < 2; ++mt) {
      int rowbase = wv * 32 + mt * 16;
      BFVec Av;   // A[row][k] = hbuf[row*32+k]; K=32 exact
      {
        const uint4* ah = (const uint4*)(hbuf + (rowbase + lrow) * PEC);
        uint4 a0 = ah[lo ? 0 : 1];     // K {0..7 | 8..15}
        uint4 a1 = ah[lo ? 2 : 3];     // K {16..23 | 24..31}
        Av.u[0] = a0.x; Av.u[1] = a0.y; Av.u[2] = a0.z; Av.u[3] = a0.w;
        Av.u[4] = a1.x; Av.u[5] = a1.y; Av.u[6] = a1.z; Av.u[7] = a1.w;
      }
#pragma unroll
      for (int nt = 0; nt < 2; ++nt) {
        BFVec Bv;  // B[k][col] = peW1T[col*32 + k]
        {
          const uint4* bp = (const uint4*)(peW1T + (nt * 16 + lrow) * PEC + (lo ? 0 : 8));
          uint4 b0 = bp[0], b1 = bp[2];
          Bv.u[0] = b0.x; Bv.u[1] = b0.y; Bv.u[2] = b0.z; Bv.u[3] = b0.w;
          Bv.u[4] = b1.x; Bv.u[5] = b1.y; Bv.u[6] = b1.z; Bv.u[7] = b1.w;
        }
        v8f acc;
#pragma unroll
        for (int i = 0; i < 8; i++) acc[i] = 0.f;
        acc = __builtin_amdgcn_wmma_f32_16x16x32_bf16(false, Av.v, false, Bv.v,
                                                      (short)0, acc, false, false);
#pragma unroll
        for (int r = 0; r < 8; r++)
          ybuf[(rowbase + rb + r) * PEC + nt * 16 + lrow] = acc[r];
      }
    }
    wn_layer_wmma(wbuf, zbuf, wnW1T, wv, lane);
  }
  __syncthreads();

  // ===== phase C: per-pair LN on PE-L1 -> sPeT ; LN+act on WN-L1 -> wbuf =====
  {
    float y[PEC];
    const float4* yb = (const float4*)(ybuf + pair * PEC);
#pragma unroll
    for (int j = 0; j < 8; ++j) {
      float4 q = yb[j];
      y[4 * j + 0] = q.x + pe_b1[4 * j + 0];
      y[4 * j + 1] = q.y + pe_b1[4 * j + 1];
      y[4 * j + 2] = q.z + pe_b1[4 * j + 2];
      y[4 * j + 3] = q.w + pe_b1[4 * j + 3];
    }
    lnact<PEC>(y, pe_g1, pe_be1, false);
#pragma unroll
    for (int o = 0; o < PEC; o++) sPeT[(p * PEC + o) * KNN + kk] = f2bf(y[o]);  // [p][c][k]

    float z[MID];
    const float4* zb = (const float4*)(zbuf + pair * MID);
#pragma unroll
    for (int j = 0; j < 4; ++j) {
      float4 q = zb[j];
      z[4 * j + 0] = q.x + wn_b1[4 * j + 0];
      z[4 * j + 1] = q.y + wn_b1[4 * j + 1];
      z[4 * j + 2] = q.z + wn_b1[4 * j + 2];
      z[4 * j + 3] = q.w + wn_b1[4 * j + 3];
    }
    lnact<MID>(z, wn_g1, wn_be1, true);
    uint4* wv4 = (uint4*)(wbuf + pair * MID);
#pragma unroll
    for (int j = 0; j < 2; ++j) {
      uint4 v;
      v.x = pk2bf(z[8 * j + 0], z[8 * j + 1]);
      v.y = pk2bf(z[8 * j + 2], z[8 * j + 3]);
      v.z = pk2bf(z[8 * j + 4], z[8 * j + 5]);
      v.w = pk2bf(z[8 * j + 6], z[8 * j + 7]);
      wv4[j] = v;
    }
  }
  __syncthreads();

  // ===== phase D: WN-L2 via WMMA =====
  wn_layer_wmma(wbuf, zbuf, wnW2T, wv, lane);
  __syncthreads();

  // ===== phase E: per-pair LN (no act) -> sWtsT =====
  {
    float z[MID];
    const float4* zb = (const float4*)(zbuf + pair * MID);
#pragma unroll
    for (int j = 0; j < 4; ++j) {
      float4 q = zb[j];
      z[4 * j + 0] = q.x + wn_b2[4 * j + 0];
      z[4 * j + 1] = q.y + wn_b2[4 * j + 1];
      z[4 * j + 2] = q.z + wn_b2[4 * j + 2];
      z[4 * j + 3] = q.w + wn_b2[4 * j + 3];
    }
    lnact<MID>(z, wn_g2, wn_be2, false);
#pragma unroll
    for (int o = 0; o < MID; o++) sWtsT[(p * MID + o) * KNN + kk] = f2bf(z[o]);  // [p][m][k]
  }
  __syncthreads();

  // ===== stage 2: outer[p][c][m] = sum_k feat[p][k][c]*wts[p][k][m] (K padded 16->32)
  {
    const int kb0 = lo ? 0 : 8;
#pragma unroll 1
    for (int pp = 0; pp < 2; ++pp) {
      int pt = wv * 2 + pp;
      BFVec Bv;
      {
        const uint4* wp4 = (const uint4*)(sWtsT + (pt * MID + lrow) * KNN);
        uint4 q0 = wp4[0], q1 = wp4[1];
        Bv.u[0] = lo ? q0.x : q1.x; Bv.u[1] = lo ? q0.y : q1.y;
        Bv.u[2] = lo ? q0.z : q1.z; Bv.u[3] = lo ? q0.w : q1.w;
        Bv.u[4] = 0u; Bv.u[5] = 0u;
        Bv.u[6] = 0u; Bv.u[7] = 0u;
      }
#pragma unroll 1
      for (int ct = 0; ct < CFEAT / 16; ++ct) {   // 10 c-tiles
        BFVec Av;
        if (ct < CIN / 16) {                       // uniform branch: gathered features
          int c = ct * 16 + lrow;
#pragma unroll
          for (int j = 0; j < 4; j++) {
            int k0 = kb0 + 2 * j;
            unsigned a0 = featsBF[(unsigned)sNei[pt * KNN + k0] * CIN + c];
            unsigned a1 = featsBF[(unsigned)sNei[pt * KNN + k0 + 1] * CIN + c];
            Av.u[j] = a0 | (a1 << 16);
          }
        } else {                                   // PE channels from LDS (transposed)
          int cc = ct * 16 + lrow - CIN;
          const uint4* pp4 = (const uint4*)(sPeT + (pt * PEC + cc) * KNN);
          uint4 av = pp4[kb0 >> 3];
          Av.u[0] = av.x; Av.u[1] = av.y; Av.u[2] = av.z; Av.u[3] = av.w;
        }
        Av.u[4] = Av.u[5] = Av.u[6] = Av.u[7] = 0u;
        v8f acc;
#pragma unroll
        for (int i = 0; i < 8; i++) acc[i] = 0.f;
        acc = __builtin_amdgcn_wmma_f32_16x16x32_bf16(false, Av.v, false, Bv.v,
                                                      (short)0, acc, false, false);
        int rbase = lo ? 0 : 8;
#pragma unroll
        for (int r = 0; r < 8; r++) {
          int cfull = ct * 16 + rbase + r;
          sOuter[pt * KTOT + cfull * MID + lrow] = f2bf(acc[r]);  // flatten c*16+m
        }
      }
    }
  }
  __syncthreads();

  // ===== stage 3: out[16p,128co] = A[16p,2560] x W[2560,128], B via TDM double-buffer
  {
    const int arow = lane & 15;
    const int akoff = lo ? 0 : 16;
    const int bcolL = lane & 15;
    const int bkoAdd = lo ? 0 : 16;
    const int bcol = wv * 16 + bcolL;
    v8f acc;
#pragma unroll
    for (int i = 0; i < 8; i++) acc[i] = 0.f;
    const unsigned* aup = (const unsigned*)(sOuter + arow * KTOT);
#pragma unroll 1
    for (int ch = 0; ch < NCH; ++ch) {
      __builtin_amdgcn_s_wait_tensorcnt(1);     // chunk ch landed in buf[ch&1]
      const unsigned short* bu =
          (const unsigned short*)(smem_raw + bbase + (unsigned)(ch & 1) * 2048u);
#pragma unroll
      for (int s = 0; s < 2; ++s) {
        int kb = ch * KCH + s * 32;
        BFVec Av, Bv;
        const uint4* ap = (const uint4*)(aup + ((kb + akoff) >> 1));
        uint4 a0 = ap[0], a1 = ap[1];
        Av.u[0] = a0.x; Av.u[1] = a0.y; Av.u[2] = a0.z; Av.u[3] = a0.w;
        Av.u[4] = a1.x; Av.u[5] = a1.y; Av.u[6] = a1.z; Av.u[7] = a1.w;
        const uint4* bp = (const uint4*)(bu + (bcolL * KCH + s * 32 + bkoAdd));
        uint4 b0 = bp[0], b1 = bp[1];
        Bv.u[0] = b0.x; Bv.u[1] = b0.y; Bv.u[2] = b0.z; Bv.u[3] = b0.w;
        Bv.u[4] = b1.x; Bv.u[5] = b1.y; Bv.u[6] = b1.z; Bv.u[7] = b1.w;
        acc = __builtin_amdgcn_wmma_f32_16x16x32_bf16(false, Av.v, false, Bv.v,
                                                      (short)0, acc, false, false);
      }
      if (ch + 2 < NCH)
        tdm_issue(bbase + (unsigned)(ch & 1) * 2048u, wcol + (ch + 2) * KCH);
    }
    int prb = lo ? 0 : 8;
#pragma unroll
    for (int r = 0; r < 8; r++)
      sOutF[(prb + r) * COUT + bcol] = acc[r] + sLin[bcol];   // bias before LN
  }
  __syncthreads();

  // ===== stage 4: LayerNorm(128) + LeakyReLU(0.1) + residual =====
  {
    int pp = tid >> 4, seg = tid & 15;
    float s = 0.f, s2 = 0.f;
#pragma unroll
    for (int j = 0; j < 8; j++) {
      float x = sOutF[pp * COUT + seg * 8 + j];
      s += x; s2 = fmaf(x, x, s2);
    }
    sRed[(pp * 16 + seg) * 2 + 0] = s;
    sRed[(pp * 16 + seg) * 2 + 1] = s2;
    __syncthreads();
    if (seg == 0) {
      float S = 0.f, S2 = 0.f;
#pragma unroll
      for (int i = 0; i < 16; i++) { S += sRed[(pp * 16 + i) * 2]; S2 += sRed[(pp * 16 + i) * 2 + 1]; }
      float mu = S * (1.f / COUT);
      float var = S2 * (1.f / COUT) - mu * mu;
      sMu[pp] = mu;
      sRs[pp] = rsqrtf(var + 1e-5f);
    }
    __syncthreads();
    int n = blockIdx.x * PTS + pp;
    if (n < Nd) {
      float mu = sMu[pp], rs = sRs[pp];
#pragma unroll
      for (int j = 0; j < 8; j++) {
        int co = seg * 8 + j;
        float x = sOutF[pp * COUT + co];
        float y = fmaf((x - mu) * rs, sLin[128 + co], sLin[256 + co]);
        y = (y >= 0.f) ? y : 0.1f * y;
        out[(size_t)n * COUT + co] = y + dfeats[(size_t)n * COUT + co];
      }
    }
  }
}

// ---------------- launcher ----------------
extern "C" void kernel_launch(void* const* d_in, const int* in_sizes, int n_in,
                              void* d_out, int out_size, void* d_ws, size_t ws_size,
                              hipStream_t stream) {
  const float* sxyz   = (const float*)d_in[0];
  const float* sfeat  = (const float*)d_in[1];
  const int*   nei    = (const int*)d_in[2];
  const float* dxyz   = (const float*)d_in[4];
  const float* dfeats = (const float*)d_in[6];
  const float* lin_w  = (const float*)d_in[27];
  const int Nd  = in_sizes[4] / 3;
  const int nsf = in_sizes[1];

  unsigned short* WtBF    = (unsigned short*)d_ws;               // 128x2560 bf16 col-major
  unsigned short* featsBF = WtBF + (size_t)KTOT * COUT;          // 10000x128 bf16
  unsigned short* smallW  = featsBF + (size_t)nsf;               // peW1T|wnW1T|wnW2T
  unsigned short* peW1T   = smallW;
  unsigned short* wnW1T   = smallW + 1024;
  unsigned short* wnW2T   = smallW + 1280;

  cvt_linw<<<(KTOT * COUT + 255) / 256, 256, 0, stream>>>(lin_w, WtBF);
  cvt_bf<<<(nsf + 255) / 256, 256, 0, stream>>>(sfeat, featsBF, nsf);
  cvt_small<<<6, 256, 0, stream>>>((const float*)d_in[11], (const float*)d_in[19],
                                   (const float*)d_in[23], smallW);

  dim3 grid((Nd + PTS - 1) / PTS);
  pointconv_main<<<grid, 256, SMEM_BYTES, stream>>>(
      sxyz, nei, dxyz, dfeats, featsBF, WtBF, peW1T, wnW1T, wnW2T,
      (const float*)d_in[7],  (const float*)d_in[8],  (const float*)d_in[9],  (const float*)d_in[10],
      (const float*)d_in[12], (const float*)d_in[13], (const float*)d_in[14],
      (const float*)d_in[15], (const float*)d_in[16], (const float*)d_in[17], (const float*)d_in[18],
      (const float*)d_in[20], (const float*)d_in[21], (const float*)d_in[22],
      (const float*)d_in[24], (const float*)d_in[25], (const float*)d_in[26],
      (const float*)d_in[28], (const float*)d_in[29], (const float*)d_in[30],
      (float*)d_out, Nd);
}
